// CharacterLevelCNN_88519275971139
// MI455X (gfx1250) — matrix operand order repaired and checked by reading, not compiled, not run on hardware
//
#include <hip/hip_runtime.h>

typedef __bf16 bf16_t;
typedef __attribute__((ext_vector_type(16))) __bf16 v16bf;
typedef __attribute__((ext_vector_type(8)))  float  v8f;
typedef __attribute__((ext_vector_type(4)))  unsigned int tdm_v4u;
typedef __attribute__((ext_vector_type(8)))  int          tdm_v8i;
typedef __attribute__((ext_vector_type(4)))  int          tdm_v4i;

#define CDIV(a, b) (((a) + (b) - 1) / (b))

#if __has_builtin(__builtin_amdgcn_tensor_load_to_lds)
#define USE_TDM 1
#else
#define USE_TDM 0
#endif

// ---------------------------------------------------------------------------
// TDM 2D tile load: rows x cols of 2-byte elements, row stride in elements.
// D# per CDNA5 ISA ch.8: group0 = {count, lds_addr, global_addr, type=2},
// group1 = {data_size=1(2B), tensor_dim0/1, tile_dim0/1, tensor_dim0_stride}.
// OOB (col >= tdim0 or row >= tdim1) zero-fills LDS -- used for K padding and
// row clamping. groups 2/3 zero (<=2D tensor). This toolchain's builtin is the
// 6-arg form: (v4u g0, v8i g1, v4i g2, v4i g3, v8i extra, i32 cpol).
// ---------------------------------------------------------------------------
#if USE_TDM
__device__ __forceinline__ void tdm_load_2d(unsigned lds_off, const void* gaddr,
                                            unsigned tdim0, unsigned tdim1,
                                            unsigned tile0, unsigned tile1,
                                            unsigned long long stride0_elems) {
    unsigned long long ga = (unsigned long long)(size_t)gaddr;
    tdm_v4u g0;
    g0[0] = 1u;                                            // count=1, user D#
    g0[1] = lds_off;                                       // LDS byte address
    g0[2] = (unsigned)(ga & 0xffffffffull);                // global_addr[31:0]
    g0[3] = (unsigned)((ga >> 32) & 0x1ffffffull) | (2u << 30);  // [56:32] | type=2
    tdm_v8i g1;
    g1[0] = (int)(1u << 16);                               // wg_mask=0, data_size=1 (2B)
    g1[1] = (int)((tdim0 & 0xffffu) << 16);                // tensor_dim0[15:0]
    g1[2] = (int)((tdim0 >> 16) | ((tdim1 & 0xffffu) << 16));
    g1[3] = (int)((tdim1 >> 16) | (tile0 << 16));          // tile_dim0
    g1[4] = (int)(tile1 & 0xffffu);                        // tile_dim1 (tile_dim2=0)
    g1[5] = (int)(stride0_elems & 0xffffffffull);          // tensor_dim0_stride lo
    g1[6] = (int)((stride0_elems >> 32) & 0xffffull);      // stride hi (dim1_stride=0)
    g1[7] = 0;
    tdm_v4i gz4;
    gz4[0] = 0; gz4[1] = 0; gz4[2] = 0; gz4[3] = 0;
    tdm_v8i gz8;
    gz8[0] = 0; gz8[1] = 0; gz8[2] = 0; gz8[3] = 0;
    gz8[4] = 0; gz8[5] = 0; gz8[6] = 0; gz8[7] = 0;
    __builtin_amdgcn_tensor_load_to_lds(g0, g1, gz4, gz4, gz8, 0);
}
#endif

// ---------------------------------------------------------------------------
// argmax over 70 chars -> embedding gather -> bf16 act0[b][l][70] (38 zero pad)
// ---------------------------------------------------------------------------
__global__ void embed_kernel(const float* __restrict__ x, const float* __restrict__ emb,
                             bf16_t* __restrict__ out, int total) {
    int i = blockIdx.x * blockDim.x + threadIdx.x;
    if (i >= total) return;
    const float* row = x + (size_t)i * 70;
    int best = 0; float bv = row[0];
    for (int c = 1; c < 70; ++c) { float v = row[c]; if (v > bv) { bv = v; best = c; } }
    bf16_t* o = out + (size_t)i * 70;
    const float* e = emb + best * 32;
    for (int c = 0; c < 32; ++c) o[c] = (bf16_t)e[c];
    for (int c = 32; c < 70; ++c) o[c] = (bf16_t)0.0f;
}

// ---------------------------------------------------------------------------
// conv weight prep: w[oc][ic][k] f32 -> wp[oc][k*Cin+ic] bf16, zero pad to Kdim
// ---------------------------------------------------------------------------
__global__ void prep_conv_w(const float* __restrict__ w, bf16_t* __restrict__ wp,
                            int Cin, int K, int Kdim, int total) {
    int j = blockIdx.x * blockDim.x + threadIdx.x;
    if (j >= total) return;
    int oc = j / Kdim, r = j % Kdim;
    bf16_t v = (bf16_t)0.0f;
    if (r < K * Cin) {
        int k = r / Cin, ic = r % Cin;
        v = (bf16_t)w[(size_t)oc * Cin * K + (size_t)ic * K + k];
    }
    wp[j] = v;
}

// ---------------------------------------------------------------------------
// Implicit-GEMM conv, fused bias+ReLU+maxpool(POOL). Channels-last act layout
// makes the im2col entry [j][p] = xflat[p*Cin + j], so both the activation
// strip (POOL*16 rows x 128 K, stride Cin) and the weight tile (128 couts x
// 128 K, stride kdim) are plain 2D tiles -> staged into LDS by the TDM.
// Block = 256 threads = 8 waves x 16 couts; grid = (pTiles, 8, B).
// Per barrier period: 2 tensor_load_to_lds + s_wait_tensorcnt, then 4 K-sub-
// chunks x POOL WMMAs fed from LDS.
// ---------------------------------------------------------------------------
template <int POOL>
__global__ void __launch_bounds__(256)
conv_wmma(const bf16_t* __restrict__ x, const bf16_t* __restrict__ wp,
          const float* __restrict__ bias, bf16_t* __restrict__ y,
          int Cin, int Lin, int Lpool, int kdim, int kvalid) {
    __shared__ bf16_t wlds[128 * 128];          // weight tile  (32 KB)
    __shared__ bf16_t stage[POOL * 16 * 128];   // im2col strip (<=12 KB)
    const int lane = threadIdx.x & 31;
    const int wave = threadIdx.x >> 5;
    const int half = lane >> 4;
    const int n    = lane & 15;
    const int b    = blockIdx.z;
    const int p0   = blockIdx.x * 16 * POOL;    // first input position of tile
    const int cb   = blockIdx.y * 128;          // block's cout base
    const int coutBase = cb + wave * 16;
    const size_t xrow0 = ((size_t)b * Lin + p0) * Cin;

    int rowsAvail = Lin - p0;
    if (rowsAvail > POOL * 16) rowsAvail = POOL * 16;
    if (rowsAvail < 0) rowsAvail = 0;

    v8f acc[POOL];
    for (int s = 0; s < POOL; ++s)
        for (int r = 0; r < 8; ++r) acc[s][r] = 0.0f;

#if USE_TDM
    const unsigned stage_off = (unsigned)(unsigned long long)(size_t)&stage[0];
    const unsigned wlds_off  = (unsigned)(unsigned long long)(size_t)&wlds[0];
#endif

    for (int kk = 0; kk < kdim; kk += 128) {
        __syncthreads();  // previous stage fully consumed
#if USE_TDM
        if (wave == 0) {
            int av = kvalid - kk;            // conv1: K-pad 490->512 zero-fills
            if (av > 128) av = 128;
            if (av < 0) av = 0;
            tdm_load_2d(stage_off, (const void*)(x + xrow0 + kk),
                        (unsigned)av, (unsigned)rowsAvail,
                        128u, (unsigned)(POOL * 16), (unsigned long long)Cin);
            tdm_load_2d(wlds_off, (const void*)(wp + (size_t)cb * kdim + kk),
                        128u, 128u, 128u, 128u, (unsigned long long)kdim);
            __builtin_amdgcn_s_wait_tensorcnt(0);
        }
#else
        for (int i = threadIdx.x; i < POOL * 16 * 128; i += 256) {
            int r = i >> 7, c = i & 127;
            int jj = kk + c;
            float v = 0.0f;
            if (jj < kvalid && r < rowsAvail) v = (float)x[xrow0 + (size_t)r * Cin + jj];
            stage[i] = (bf16_t)v;
        }
        for (int i = threadIdx.x; i < 128 * 128; i += 256) {
            int r = i >> 7, c = i & 127;
            wlds[i] = wp[(size_t)(cb + r) * kdim + kk + c];
        }
#endif
        __syncthreads();  // stage ready for all 8 waves

#pragma unroll
        for (int k2 = 0; k2 < 4; ++k2) {
            // A fragment (16x32 bf16): lane row = wave*16+n, K = k2*32 +
            // {half*16..+15}
            v16bf afrag;
            const bf16_t* wa = &wlds[(wave * 16 + n) * 128 + k2 * 32 + half * 16];
#pragma unroll
            for (int e = 0; e < 8; ++e) afrag[e] = wa[e];
#pragma unroll
            for (int e = 0; e < 8; ++e) afrag[8 + e] = wa[8 + e];

#pragma unroll
            for (int s = 0; s < POOL; ++s) {
                // B fragment (32x16): column n holds 16 consecutive K at half*16;
                // subtile s column n <-> input position p0 + POOL*n + s.
                const int row = (POOL == 1) ? n : (3 * n + s);
                const bf16_t* bp = &stage[row * 128 + k2 * 32 + half * 16];
                v16bf bfrag;
#pragma unroll
                for (int e = 0; e < 16; ++e) bfrag[e] = bp[e];
                acc[s] = __builtin_amdgcn_wmma_f32_16x16x32_bf16(
                    false, afrag, false, bfrag, (short)0, acc[s], false, false);
            }
        }
    }

    // epilogue: max over pool taps, + bias, ReLU, store bf16 channels-last
    const int gp = blockIdx.x * 16 + n;  // pooled position
    if (gp < Lpool) {
#pragma unroll
        for (int r = 0; r < 8; ++r) {
            const int cout = coutBase + r + 8 * half;  // C/D layout: M = r + 8*half
            float v = acc[0][r];
#pragma unroll
            for (int s = 1; s < POOL; ++s) v = fmaxf(v, acc[s][r]);
            v += bias[cout];
            v = fmaxf(v, 0.0f);
            y[((size_t)b * Lpool + gp) * 1024 + cout] = (bf16_t)v;
        }
    }
}

// ---------------------------------------------------------------------------
// FC: y[m][n] = act(x[m]·w[n] + bias[n]). x bf16 [M][K], w fp32 [N][K] row-major
// (weights converted to bf16 in-registers — weight-streaming GEMM, M=64).
// One wave per 16x16 tile; grid = (ceil(N/16), M/16), block = 32.
// ---------------------------------------------------------------------------
template <bool RELU, typename OUT>
__global__ void __launch_bounds__(32)
fc_wmma(const bf16_t* __restrict__ x, const float* __restrict__ w,
        const float* __restrict__ bias, OUT* __restrict__ out, int N, int K) {
    const int lane = threadIdx.x & 31;
    const int half = lane >> 4;
    const int n16  = lane & 15;
    const int nBase = blockIdx.x * 16;
    const int mBase = blockIdx.y * 16;
    const int nn = nBase + n16;
    const int m  = mBase + n16;

    v8f acc;
    for (int r = 0; r < 8; ++r) acc[r] = 0.0f;

    for (int kk = 0; kk < K; kk += 32) {
        v16bf afrag;
        const bf16_t* xa = x + (size_t)m * K + kk + half * 16;
#pragma unroll
        for (int e = 0; e < 8; ++e) afrag[e] = xa[e];
#pragma unroll
        for (int e = 0; e < 8; ++e) afrag[8 + e] = xa[8 + e];

        v16bf bfrag;
        if (nn < N) {
            const float* wb = w + (size_t)nn * K + kk + half * 16;
            __builtin_prefetch(wb + 32, 0, 0);  // next K-chunk -> global_prefetch_b8
#pragma unroll
            for (int e = 0; e < 16; ++e) bfrag[e] = (bf16_t)wb[e];
        } else {
#pragma unroll
            for (int e = 0; e < 16; ++e) bfrag[e] = (bf16_t)0.0f;
        }
        acc = __builtin_amdgcn_wmma_f32_16x16x32_bf16(
            false, afrag, false, bfrag, (short)0, acc, false, false);
    }

    if (nn < N) {
        float bv = bias[nn];
#pragma unroll
        for (int r = 0; r < 8; ++r) {
            int mm = mBase + r + 8 * half;
            float v = acc[r] + bv;
            if (RELU) v = fmaxf(v, 0.0f);
            out[(size_t)mm * N + nn] = (OUT)v;
        }
    }
}

// ---------------------------------------------------------------------------
// FC1: act layout is [b][l][c] (j = l*1024 + c) but fw1 is indexed by c*34 + l
// (NCH flatten). Absorb the permutation into the B-fragment gather (stride 34).
// ---------------------------------------------------------------------------
__global__ void __launch_bounds__(32)
fc1_wmma(const bf16_t* __restrict__ x, const float* __restrict__ w,
         const float* __restrict__ bias, bf16_t* __restrict__ out, int N, int K) {
    const int lane = threadIdx.x & 31;
    const int half = lane >> 4;
    const int n16  = lane & 15;
    const int nBase = blockIdx.x * 16;
    const int mBase = blockIdx.y * 16;
    const int nn = nBase + n16;
    const int m  = mBase + n16;

    v8f acc;
    for (int r = 0; r < 8; ++r) acc[r] = 0.0f;

    for (int kk = 0; kk < K; kk += 32) {
        v16bf afrag;
        const bf16_t* xa = x + (size_t)m * K + kk + half * 16;
#pragma unroll
        for (int e = 0; e < 8; ++e) afrag[e] = xa[e];
#pragma unroll
        for (int e = 0; e < 8; ++e) afrag[8 + e] = xa[8 + e];

        // j = kk + half*16 + e ; l = j/1024, c = j%1024 (16-run stays in one l block)
        const int j0 = kk + half * 16;
        const int l  = j0 >> 10;
        const int c0 = j0 & 1023;
        v16bf bfrag;
        if (nn < N) {
            const float* wb = w + (size_t)nn * K + (size_t)c0 * 34 + l;
#pragma unroll
            for (int e = 0; e < 16; ++e) bfrag[e] = (bf16_t)wb[e * 34];
        } else {
#pragma unroll
            for (int e = 0; e < 16; ++e) bfrag[e] = (bf16_t)0.0f;
        }
        acc = __builtin_amdgcn_wmma_f32_16x16x32_bf16(
            false, afrag, false, bfrag, (short)0, acc, false, false);
    }

    if (nn < N) {
        float bv = bias[nn];
#pragma unroll
        for (int r = 0; r < 8; ++r) {
            int mm = mBase + r + 8 * half;
            out[(size_t)mm * N + nn] = (bf16_t)fmaxf(acc[r] + bv, 0.0f);
        }
    }
}

// ---------------------------------------------------------------------------
extern "C" void kernel_launch(void* const* d_in, const int* in_sizes, int n_in,
                              void* d_out, int out_size, void* d_ws, size_t ws_size,
                              hipStream_t stream) {
    (void)in_sizes; (void)n_in; (void)out_size; (void)ws_size;
    const float* x   = (const float*)d_in[0];
    const float* emb = (const float*)d_in[1];
    const float* cw[6] = {(const float*)d_in[2],  (const float*)d_in[4],
                          (const float*)d_in[6],  (const float*)d_in[8],
                          (const float*)d_in[10], (const float*)d_in[12]};
    const float* cb[6] = {(const float*)d_in[3],  (const float*)d_in[5],
                          (const float*)d_in[7],  (const float*)d_in[9],
                          (const float*)d_in[11], (const float*)d_in[13]};
    const float* fw1 = (const float*)d_in[14]; const float* fb1 = (const float*)d_in[15];
    const float* fw2 = (const float*)d_in[16]; const float* fb2 = (const float*)d_in[17];
    const float* fw3 = (const float*)d_in[18]; const float* fb3 = (const float*)d_in[19];

    char* ws = (char*)d_ws;
    size_t off = 0;
    auto balloc = [&](size_t elems) -> bf16_t* {
        bf16_t* p = (bf16_t*)(ws + off);
        // +64KB slack: TDM tile rows may legitimately read past a row's tail
        off += ((elems * sizeof(bf16_t) + 255) & ~(size_t)255) + 65536;
        return p;
    };
    bf16_t* act0 = balloc((size_t)64 * 1014 * 70);   // embedded input, channels-last
    bf16_t* bufA = balloc((size_t)64 * 336 * 1024);  // acts 1,3,5
    bf16_t* bufB = balloc((size_t)64 * 110 * 1024);  // acts 2,4,6
    bf16_t* wp1  = balloc((size_t)1024 * 512);
    bf16_t* wp2  = balloc((size_t)1024 * 7168);
    bf16_t* wp3  = balloc((size_t)1024 * 3072);
    bf16_t* wp4  = balloc((size_t)1024 * 3072);
    bf16_t* wp5  = balloc((size_t)1024 * 3072);
    bf16_t* wp6  = balloc((size_t)1024 * 3072);
    bf16_t* fc1o = balloc((size_t)64 * 2048);
    bf16_t* fc2o = balloc((size_t)64 * 2048);

    // --- prep: embedding + conv weight reorder/convert -----------------------
    embed_kernel<<<CDIV(64 * 1014, 256), 256, 0, stream>>>(x, emb, act0, 64 * 1014);
    prep_conv_w<<<CDIV(1024 * 512, 256),  256, 0, stream>>>(cw[0], wp1,   70, 7,  512, 1024 * 512);
    prep_conv_w<<<CDIV(1024 * 7168, 256), 256, 0, stream>>>(cw[1], wp2, 1024, 7, 7168, 1024 * 7168);
    prep_conv_w<<<CDIV(1024 * 3072, 256), 256, 0, stream>>>(cw[2], wp3, 1024, 3, 3072, 1024 * 3072);
    prep_conv_w<<<CDIV(1024 * 3072, 256), 256, 0, stream>>>(cw[3], wp4, 1024, 3, 3072, 1024 * 3072);
    prep_conv_w<<<CDIV(1024 * 3072, 256), 256, 0, stream>>>(cw[4], wp5, 1024, 3, 3072, 1024 * 3072);
    prep_conv_w<<<CDIV(1024 * 3072, 256), 256, 0, stream>>>(cw[5], wp6, 1024, 3, 3072, 1024 * 3072);

    // --- conv stack (bias+ReLU[+pool3] fused) --------------------------------
    // conv1: 70->1024 k7, L 1014->1008, pool3 -> 336 (Kdim 490 padded to 512)
    conv_wmma<3><<<dim3(21, 8, 64), 256, 0, stream>>>(act0, wp1, cb[0], bufA,   70, 1014, 336,  512,  490);
    // conv2: k7, L 336->330, pool3 -> 110
    conv_wmma<3><<<dim3(7, 8, 64),  256, 0, stream>>>(bufA, wp2, cb[1], bufB, 1024,  336, 110, 7168, 7168);
    // conv3: k3, L 110->108
    conv_wmma<1><<<dim3(7, 8, 64),  256, 0, stream>>>(bufB, wp3, cb[2], bufA, 1024,  110, 108, 3072, 3072);
    // conv4: k3, L 108->106
    conv_wmma<1><<<dim3(7, 8, 64),  256, 0, stream>>>(bufA, wp4, cb[3], bufB, 1024,  108, 106, 3072, 3072);
    // conv5: k3, L 106->104
    conv_wmma<1><<<dim3(7, 8, 64),  256, 0, stream>>>(bufB, wp5, cb[4], bufA, 1024,  106, 104, 3072, 3072);
    // conv6: k3, L 104->102, pool3 -> 34
    conv_wmma<3><<<dim3(3, 8, 64),  256, 0, stream>>>(bufA, wp6, cb[5], bufB, 1024,  104,  34, 3072, 3072);

    // --- FC head -------------------------------------------------------------
    fc1_wmma<<<dim3(128, 4), 32, 0, stream>>>(bufB, fw1, fb1, fc1o, 2048, 34816);
    fc_wmma<true,  bf16_t><<<dim3(128, 4), 32, 0, stream>>>(fc1o, fw2, fb2, fc2o, 2048, 2048);
    fc_wmma<false, float ><<<dim3(1, 4),   32, 0, stream>>>(fc2o, fw3, fb3, (float*)d_out, 14, 2048);
}
